// LowRankAttention_78194174591584
// MI455X (gfx1250) — matrix operand order repaired and checked by reading, not compiled, not run on hardware
//
#include <hip/hip_runtime.h>
#include <hip/hip_bf16.h>

// ---------------------------------------------------------------------------
// LowRankAttention on MI455X (gfx1250, wave32).
// All GEMMs in bf16 WMMA (v_wmma_f32_16x16x32_bf16), fp32 accumulate.
// Attention is fused flash-style (online softmax); K/V^T tiles are staged
// into LDS by the Tensor Data Mover (tensor_load_to_lds, TENSORcnt) with
// double buffering so DMA overlaps WMMA issue.
// ---------------------------------------------------------------------------

#define DIMN   1024
#define RANK   256
#define BATCH  8
#define SEQ    2048
#define MTOT   (BATCH * SEQ)   // 16384

typedef __attribute__((ext_vector_type(16))) __bf16 v16bf;
typedef __attribute__((ext_vector_type(8)))  __bf16 v8bf;
typedef __attribute__((ext_vector_type(8)))  float  v8f;
typedef __attribute__((ext_vector_type(4)))  unsigned int v4u32;
typedef __attribute__((ext_vector_type(8)))  int  v8i32;
typedef __attribute__((ext_vector_type(4)))  int  v4i32;

#if defined(__has_builtin)
#  if __has_builtin(__builtin_amdgcn_tensor_load_to_lds)
#    define HAVE_TDM 1
#  else
#    define HAVE_TDM 0
#  endif
#else
#  define HAVE_TDM 0
#endif

// --- fragment helpers -------------------------------------------------------

__device__ inline v16bf join8(v8bf lo, v8bf hi) {
    v16bf r;
#pragma unroll
    for (int i = 0; i < 8; ++i) { r[i] = lo[i]; r[i + 8] = hi[i]; }
    return r;
}

// 16 contiguous bf16 (two b128 loads)
__device__ inline v16bf load_bf16x16(const __bf16* p) {
    return join8(*(const v8bf*)p, *(const v8bf*)(p + 8));
}

// two disjoint groups of 8 contiguous bf16 (A-fragment K-interleave)
__device__ inline v16bf load_bf16x8x2(const __bf16* p0, const __bf16* p1) {
    return join8(*(const v8bf*)p0, *(const v8bf*)p1);
}

__device__ inline v8f wmma_bf16(v16bf a, v16bf b, v8f c) {
    // D = A(16x32) * B(32x16) + C(16x16), fp32 accumulate
    return __builtin_amdgcn_wmma_f32_16x16x32_bf16(
        /*neg_a=*/false, a, /*neg_b=*/false, b,
        /*c_mod=*/(short)0, c, /*reuse_a=*/false, /*reuse_b=*/false);
}

// half-wave (16-lane) row reductions; rows of a WMMA C tile live across the
// 16 lanes of each wave half, so xor masks 1/2/4/8 stay inside the half.
__device__ inline float hmax16(float x) {
    x = fmaxf(x, __shfl_xor(x, 1, 32));
    x = fmaxf(x, __shfl_xor(x, 2, 32));
    x = fmaxf(x, __shfl_xor(x, 4, 32));
    x = fmaxf(x, __shfl_xor(x, 8, 32));
    return x;
}
__device__ inline float hsum16(float x) {
    x += __shfl_xor(x, 1, 32);
    x += __shfl_xor(x, 2, 32);
    x += __shfl_xor(x, 4, 32);
    x += __shfl_xor(x, 8, 32);
    return x;
}

// --- TDM descriptor helpers (CDNA5 ISA ch. 8, D# groups 0/1) ----------------
#if HAVE_TDM
// Group 0: count=1 | lds_addr | global_addr[56:0] | type=2 ("image")
__device__ inline v4u32 tdm_g0(unsigned ldsAddr, const void* gptr) {
    const unsigned long long ga = (unsigned long long)(size_t)gptr;
    v4u32 g0;
    g0[0] = 1u;                                        // count=1, no gather
    g0[1] = ldsAddr;                                   // LDS byte address
    g0[2] = (unsigned)(ga & 0xffffffffu);              // global_addr[31:0]
    g0[3] = (unsigned)((ga >> 32) & 0x01ffffffu)       // global_addr[56:32]
          | 0x80000000u;                               // type=2 in bits 127:126
    return g0;
}
// Group 1 for a 2-D bf16 tensor: data_size=2B, dims/tiles/stride0.
__device__ inline v8i32 tdm_g1_2d(unsigned dim0, unsigned dim1,
                                  unsigned tile0, unsigned tile1,
                                  unsigned stride0) {
    v8i32 g1 = {};
    g1[0] = (int)(1u << 16);                           // data_size=1 -> 2 bytes
    g1[1] = (int)(dim0 << 16);                         // tensor_dim0[15:0]
    g1[2] = (int)((dim0 >> 16) | (dim1 << 16));        // dim0 hi | dim1 lo
    g1[3] = (int)((dim1 >> 16) | (tile0 << 16));       // dim1 hi | tile_dim0
    g1[4] = (int)(tile1 & 0xffffu);                    // tile_dim1 (tile_dim2=0)
    g1[5] = (int)stride0;                              // tensor_dim0_stride lo
    g1[6] = 0;                                         // stride0 hi | stride1 lo
    g1[7] = 0;
    return g1;
}
__device__ inline void tdm_load(v4u32 g0, v8i32 g1) {
    v4i32 z4 = {};
#if __clang_major__ >= 23
    v8i32 z8 = {};
    __builtin_amdgcn_tensor_load_to_lds(g0, g1, z4, z4, z8, 0);
#else
    __builtin_amdgcn_tensor_load_to_lds(g0, g1, z4, z4, 0);
#endif
}
// low 32 bits of a generic shared pointer == LDS byte offset (aperture map)
__device__ inline unsigned lds_addr32(const void* p) {
    return (unsigned)(size_t)p;
}
#endif

// --- kernel 0: weight fp32 -> bf16 -----------------------------------------

__global__ void cvt_f32_bf16(const float* __restrict__ src,
                             __bf16* __restrict__ dst, int n) {
    int i = blockIdx.x * blockDim.x + threadIdx.x;
    if (i < n) dst[i] = (__bf16)src[i];
}

// --- kernel 1: projection y = x @ W^T  (contract along DIMN) ----------------
// X: [MTOT, DIMN] fp32 (converted inline), W: [RANK, DIMN] bf16.
// dst  != null -> row-major [MTOT, RANK] bf16
// dstT != null -> transposed [BATCH, RANK, SEQ] bf16 (for V)

__global__ __launch_bounds__(128) void proj_kernel(
    const float* __restrict__ X, const __bf16* __restrict__ W,
    __bf16* __restrict__ dst, __bf16* __restrict__ dstT) {
    const int lane   = threadIdx.x & 31;
    const int wave   = threadIdx.x >> 5;
    const int laneHi = lane >> 4;
    const int ln16   = lane & 15;
    const int mBase  = (blockIdx.x * 4 + wave) * 16;
    const int nBase  = blockIdx.y * 16;

    const float*  xrow = X + (size_t)(mBase + ln16) * DIMN;
    const __bf16* wrow = W + (size_t)(nBase + ln16) * DIMN;

    v8f acc = {};
#pragma unroll 4
    for (int k0 = 0; k0 < DIMN; k0 += 32) {
        // A fragment: lane holds row m = ln16; K chunks {0..7,16..23} (+8 hi)
        const int klo = k0 + (laneHi ? 8 : 0);
        float4 f0 = *(const float4*)(xrow + klo);
        float4 f1 = *(const float4*)(xrow + klo + 4);
        float4 f2 = *(const float4*)(xrow + klo + 16);
        float4 f3 = *(const float4*)(xrow + klo + 20);
        v16bf a;
        a[0]=(__bf16)f0.x;  a[1]=(__bf16)f0.y;  a[2]=(__bf16)f0.z;  a[3]=(__bf16)f0.w;
        a[4]=(__bf16)f1.x;  a[5]=(__bf16)f1.y;  a[6]=(__bf16)f1.z;  a[7]=(__bf16)f1.w;
        a[8]=(__bf16)f2.x;  a[9]=(__bf16)f2.y;  a[10]=(__bf16)f2.z; a[11]=(__bf16)f2.w;
        a[12]=(__bf16)f3.x; a[13]=(__bf16)f3.y; a[14]=(__bf16)f3.z; a[15]=(__bf16)f3.w;
        // B fragment: lane holds column n = rank r = ln16; W row r contiguous
        v16bf b = load_bf16x8x2(wrow + k0 + (laneHi ? 8 : 0), wrow + k0 + (laneHi ? 8 : 0) + 16);
        acc = wmma_bf16(a, b, acc);
    }
#pragma unroll
    for (int v = 0; v < 8; ++v) {
        const int m = mBase + v + (laneHi ? 8 : 0);
        const int n = nBase + ln16;
        const __bf16 val = (__bf16)acc[v];
        if (dst)  dst[(size_t)m * RANK + n] = val;
        if (dstT) {
            const int b_ = m / SEQ, s = m % SEQ;
            dstT[((size_t)b_ * RANK + n) * SEQ + s] = val;
        }
    }
}

// --- kernel 2: fused flash attention over rank-256 Q/K/V --------------------
// Qb/Kb: [B*S, R] bf16, Vtb: [B, R, S] bf16, AVb out: [B*S, R] bf16.
// 256 threads = 8 waves; each wave owns a 16-row query tile. K/V^T tiles are
// DMA'd into double-buffered LDS by the TDM (wave 0 posts descriptors) so the
// tile-(i+1) transfer overlaps tile-i WMMA work.

__global__ __launch_bounds__(256) void attn_kernel(
    const __bf16* __restrict__ Qb, const __bf16* __restrict__ Kb,
    const __bf16* __restrict__ Vtb, __bf16* __restrict__ AVb) {
    __shared__ alignas(16) __bf16 kTile[2][32 * RANK];  // 2 x 16 KB  [key][r]
    __shared__ alignas(16) __bf16 vTile[2][RANK * 32];  // 2 x 16 KB  [r][key]
    __shared__ alignas(16) __bf16 pScr[8 * 16 * 32];    // 8 KB per-wave P

    const int tid    = threadIdx.x;
    const int lane   = tid & 31;
    const int wave   = tid >> 5;
    const int laneHi = lane >> 4;
    const int ln16   = lane & 15;
    const int b      = blockIdx.x >> 4;      // SEQ/128 = 16 q-groups per batch
    const int qGroup = blockIdx.x & 15;
    const int mBase  = qGroup * 128 + wave * 16;   // q row within batch

    // Preload this wave's Q tile (16 x 256) as 8 A-fragments.
    v16bf qf[8];
    {
        const __bf16* qrow = Qb + ((size_t)(b * SEQ + mBase) + ln16) * RANK;
#pragma unroll
        for (int rc = 0; rc < 8; ++rc) {
            const int klo = rc * 32 + (laneHi ? 8 : 0);
            qf[rc] = load_bf16x8x2(qrow + klo, qrow + klo + 16);
        }
    }

    v8f av[16] = {};                 // 16 rank tiles x 8 rows = 128 VGPRs
    float mrow[8], lrow[8];
#pragma unroll
    for (int v = 0; v < 8; ++v) { mrow[v] = -1e30f; lrow[v] = 0.f; }

    __bf16* pmem = pScr + wave * (16 * 32);
    const float scale = 0.0625f;     // 1/sqrt(RANK)
    const int NIT = SEQ / 32;

#if HAVE_TDM
    // Static D# group-1 descriptors for the two 2-D tensors.
    // K  : tensor [SEQ rows, RANK] row-major, tile 32 x 256 (tile0=inner dim)
    // V^T: tensor [RANK rows, SEQ] row-major, tile 256 x 32
    const v8i32 g1K = tdm_g1_2d(RANK, SEQ, RANK, 32, RANK);
    const v8i32 g1V = tdm_g1_2d(SEQ, RANK, 32, RANK, SEQ);
    const unsigned ldsK[2] = { lds_addr32(&kTile[0][0]), lds_addr32(&kTile[1][0]) };
    const unsigned ldsV[2] = { lds_addr32(&vTile[0][0]), lds_addr32(&vTile[1][0]) };
    const __bf16* kBase = Kb + (size_t)b * SEQ * RANK;
    const __bf16* vBase = Vtb + (size_t)b * RANK * SEQ;

    if (wave == 0) {                 // prologue: DMA tile 0 into buffer 0
        tdm_load(tdm_g0(ldsK[0], kBase), g1K);
        tdm_load(tdm_g0(ldsV[0], vBase), g1V);
    }
#endif

    for (int it = 0; it < NIT; ++it) {
        const int buf = it & 1;
        const int kk0 = it * 32;
        __syncthreads();             // all waves done reading buf^1
#if HAVE_TDM
        if (wave == 0) {
            if (it + 1 < NIT) {      // DMA next tile into the other buffer
                const int kn = kk0 + 32;
                tdm_load(tdm_g0(ldsK[buf ^ 1], kBase + (size_t)kn * RANK), g1K);
                tdm_load(tdm_g0(ldsV[buf ^ 1], vBase + kn), g1V);
                // per-wave tensor ops retire in order: <=2 outstanding means
                // the current tile's two descriptors have landed in LDS
                __builtin_amdgcn_s_wait_tensorcnt((short)2);
            } else {
                __builtin_amdgcn_s_wait_tensorcnt((short)0);
            }
        }
#else
        // fallback: cooperative VGPR staging (1024 x 16B per tile)
#pragma unroll
        for (int i = 0; i < 4; ++i) {
            const int idx = tid + i * 256;
            const int row = idx >> 5, c8 = idx & 31;
            *(uint4*)(&kTile[buf][0] + row * RANK + c8 * 8) =
                *(const uint4*)(Kb + (size_t)(b * SEQ + kk0 + row) * RANK + c8 * 8);
        }
#pragma unroll
        for (int i = 0; i < 4; ++i) {
            const int idx = tid + i * 256;
            const int r = idx >> 2, c8 = idx & 3;
            *(uint4*)(&vTile[buf][0] + r * 32 + c8 * 8) =
                *(const uint4*)(Vtb + ((size_t)b * RANK + r) * SEQ + kk0 + c8 * 8);
        }
#endif
        __syncthreads();             // tile `it` visible to every wave

        const __bf16* kT = &kTile[buf][0];
        const __bf16* vT = &vTile[buf][0];

        // scores: S = Q K^T, two 16x16 tiles (keys kk0..+15 and +16..+31)
        v8f sc0 = {}, sc1 = {};
#pragma unroll
        for (int rc = 0; rc < 8; ++rc) {
            const int r0 = rc * 32 + (laneHi ? 8 : 0);
            v16bf b0 = load_bf16x8x2(kT + ln16 * RANK + r0, kT + ln16 * RANK + r0 + 16);
            v16bf b1 = load_bf16x8x2(kT + (16 + ln16) * RANK + r0, kT + (16 + ln16) * RANK + r0 + 16);
            sc0 = wmma_bf16(qf[rc], b0, sc0);
            sc1 = wmma_bf16(qf[rc], b1, sc1);
        }

        // online softmax; emit P (bf16) into this wave's LDS scratch
#pragma unroll
        for (int v = 0; v < 8; ++v) {
            const float s0 = sc0[v] * scale;
            const float s1 = sc1[v] * scale;
            const float rmax = hmax16(fmaxf(s0, s1));
            const float mnew = fmaxf(mrow[v], rmax);
            const float alpha = __expf(mrow[v] - mnew);
            const float p0 = __expf(s0 - mnew);
            const float p1 = __expf(s1 - mnew);
            lrow[v] = lrow[v] * alpha + hsum16(p0 + p1);
            mrow[v] = mnew;
#pragma unroll
            for (int n = 0; n < 16; ++n) av[n][v] *= alpha;
            const int prow = v + (laneHi ? 8 : 0);
            pmem[prow * 32 + ln16]      = (__bf16)p0;
            pmem[prow * 32 + 16 + ln16] = (__bf16)p1;
        }

        // reload P in A-fragment layout (same-wave LDS ops are in-order)
        const int klo = laneHi ? 8 : 0;
        v16bf pf = load_bf16x8x2(pmem + ln16 * 32 + klo,
                                 pmem + ln16 * 32 + klo + 16);

        // AV += P(16x32) * V(32x16) per rank tile; V^T rows are contiguous
#pragma unroll
        for (int n = 0; n < 16; ++n) {
            v16bf bv = load_bf16x8x2(vT + (n * 16 + ln16) * 32 + klo, vT + (n * 16 + ln16) * 32 + klo + 16);
            av[n] = wmma_bf16(pf, bv, av[n]);
        }
    }

    // normalize rows and store AV (bf16)
#pragma unroll
    for (int v = 0; v < 8; ++v) {
        const float inv = 1.f / lrow[v];
        const int m = mBase + v + (laneHi ? 8 : 0);
        __bf16* orow = AVb + (size_t)(b * SEQ + m) * RANK;
#pragma unroll
        for (int n = 0; n < 16; ++n)
            orow[n * 16 + ln16] = (__bf16)(av[n][v] * inv);
    }
}

// --- kernel 3: output projection out = AV @ Wo^T (fp32 out) -----------------

__global__ __launch_bounds__(256) void oproj_kernel(
    const __bf16* __restrict__ AVb, const __bf16* __restrict__ Wob,
    float* __restrict__ out) {
    const int lane   = threadIdx.x & 31;
    const int wave   = threadIdx.x >> 5;
    const int laneHi = lane >> 4;
    const int ln16   = lane & 15;
    const int mBase  = (blockIdx.x * 8 + wave) * 16;
    const int nBase  = blockIdx.y * 16;

    const __bf16* arow = AVb + (size_t)(mBase + ln16) * RANK;
    const __bf16* wrow = Wob + (size_t)(nBase + ln16) * RANK;  // Wo[d, r]

    v8f acc = {};
#pragma unroll
    for (int k0 = 0; k0 < RANK; k0 += 32) {
        const int klo = k0 + (laneHi ? 8 : 0);
        v16bf a = load_bf16x8x2(arow + klo, arow + klo + 16);
        v16bf b = load_bf16x8x2(wrow + klo, wrow + klo + 16);
        acc = wmma_bf16(a, b, acc);
    }
#pragma unroll
    for (int v = 0; v < 8; ++v) {
        const int m = mBase + v + (laneHi ? 8 : 0);
        out[(size_t)m * DIMN + nBase + ln16] = acc[v];
    }
}

// --- host launcher ----------------------------------------------------------

extern "C" void kernel_launch(void* const* d_in, const int* in_sizes, int n_in,
                              void* d_out, int out_size, void* d_ws, size_t ws_size,
                              hipStream_t stream) {
    const float* q  = (const float*)d_in[0];
    const float* k  = (const float*)d_in[1];
    const float* v  = (const float*)d_in[2];
    const float* Wq = (const float*)d_in[3];
    const float* Wk = (const float*)d_in[4];
    const float* Wv = (const float*)d_in[5];
    const float* Wo = (const float*)d_in[6];
    float* out = (float*)d_out;

    // workspace carve-up (~36 MB total, bf16)
    const size_t WN = (size_t)RANK * DIMN;       // 262144 per weight
    const size_t AN = (size_t)MTOT * RANK;       // 4194304 per activation
    __bf16* Wqb = (__bf16*)d_ws;
    __bf16* Wkb = Wqb + WN;
    __bf16* Wvb = Wkb + WN;
    __bf16* Wob = Wvb + WN;
    __bf16* Qb  = Wob + WN;
    __bf16* Kb  = Qb + AN;
    __bf16* Vtb = Kb + AN;      // [BATCH, RANK, SEQ]
    __bf16* AVb = Vtb + AN;
    (void)in_sizes; (void)n_in; (void)out_size; (void)ws_size;

    // 0) weights -> bf16
    const int nW = (int)WN;
    cvt_f32_bf16<<<(nW + 255) / 256, 256, 0, stream>>>(Wq, Wqb, nW);
    cvt_f32_bf16<<<(nW + 255) / 256, 256, 0, stream>>>(Wk, Wkb, nW);
    cvt_f32_bf16<<<(nW + 255) / 256, 256, 0, stream>>>(Wv, Wvb, nW);
    cvt_f32_bf16<<<(nW + 255) / 256, 256, 0, stream>>>(Wo, Wob, nW);

    // 1) projections (V written transposed)
    dim3 pg(MTOT / (16 * 4), RANK / 16);         // (256, 16)
    proj_kernel<<<pg, 128, 0, stream>>>(q, Wqb, Qb, nullptr);
    proj_kernel<<<pg, 128, 0, stream>>>(k, Wkb, Kb, nullptr);
    proj_kernel<<<pg, 128, 0, stream>>>(v, Wvb, nullptr, Vtb);

    // 2) fused attention (TDM double-buffered tile staging)
    attn_kernel<<<BATCH * (SEQ / 128), 256, 0, stream>>>(Qb, Kb, Vtb, AVb);

    // 3) output projection
    dim3 og(MTOT / (16 * 8), DIMN / 16);         // (128, 64)
    oproj_kernel<<<og, 256, 0, stream>>>(AVb, Wob, out);
}
